// CrossViewAttention_44521630990768
// MI455X (gfx1250) — matrix operand order repaired and checked by reading, not compiled, not run on hardware
//
#include <hip/hip_runtime.h>
#include <hip/hip_bf16.h>

typedef __attribute__((ext_vector_type(16))) __bf16 v16bf;
typedef __attribute__((ext_vector_type(8)))  __bf16 v8bf;
typedef __attribute__((ext_vector_type(8)))  float  v8f;

#define WMMA_BF16(a, b, c) \
    __builtin_amdgcn_wmma_f32_16x16x32_bf16(false, (a), false, (b), (short)0, (c), false, false)

// two contiguous 128-bit loads -> one 16-half fragment register group
__device__ __forceinline__ v16bf ld16pair(const __bf16* p0, const __bf16* p1) {
    v8bf lo = *(const v8bf*)p0;
    v8bf hi = *(const v8bf*)p1;
    return __builtin_shufflevector(lo, hi, 0, 1, 2, 3, 4, 5, 6, 7,
                                   8, 9, 10, 11, 12, 13, 14, 15);
}

// ---------------------------------------------------------------------------
#define B_     2
#define NCAM   6
#define BN     12
#define DIM_   128
#define HEADS_ 4
#define DH_    32
#define FH_    28
#define FW_    60
#define PIX    1680
#define HQ_    50
#define Q_     2500
#define FEAT   256
#define KCONV  2304        // 256*9, tap-major: k = tap*256 + cin
#define KCCH   72          // KCONV/32
#define MS     28
#define KKEYS  784
#define KCH    25          // ceil(784/32)
#define QTILES 157         // ceil(2500/16)

// ===========================================================================
// geometry: d = E_inv @ [I_inv @ p ; 1]   -> dg (bn,4,1680)
__global__ void geom_kernel(const float* __restrict__ Iinv,
                            const float* __restrict__ Einv,
                            const float* __restrict__ plane,
                            float* __restrict__ dg) {
    int idx = blockIdx.x * 256 + threadIdx.x;
    if (idx >= BN * PIX) return;
    int bn = idx / PIX, p = idx % PIX;
    float px = plane[p], py = plane[PIX + p], pz = plane[2 * PIX + p];
    const float* I = Iinv + bn * 9;
    float c0 = I[0] * px + I[1] * py + I[2] * pz;
    float c1 = I[3] * px + I[4] * py + I[5] * pz;
    float c2 = I[6] * px + I[7] * py + I[8] * pz;
    const float* E = Einv + bn * 16;
#pragma unroll
    for (int i = 0; i < 4; ++i)
        dg[((size_t)(bn * 4 + i)) * PIX + p] =
            E[i * 4 + 0] * c0 + E[i * 4 + 1] * c1 + E[i * 4 + 2] * c2 + E[i * 4 + 3];
}

__global__ void cemb_kernel(const float* __restrict__ camw,
                            const float* __restrict__ Einv,
                            float* __restrict__ cemb) {
    int bn = blockIdx.x, c = threadIdx.x;
    float acc = 0.f;
#pragma unroll
    for (int ci = 0; ci < 4; ++ci)
        acc += camw[(size_t)c * 36 + ci * 9 + 4] * Einv[(size_t)bn * 16 + ci * 4 + 3];
    cemb[bn * DIM_ + c] = acc;
}

// w_embed = conv3x3(bev_grid (2,50,50)) -> (2500,128) pixel-major
__global__ __launch_bounds__(128) void wemb_kernel(const float* __restrict__ grid,
                                                   const float* __restrict__ bw,
                                                   float* __restrict__ wemb) {
    int p = blockIdx.x, y = p / HQ_, x = p % HQ_, c = threadIdx.x;
    const float* wr = bw + (size_t)c * 18;
    float acc = 0.f;
#pragma unroll
    for (int ci = 0; ci < 2; ++ci)
#pragma unroll
        for (int ky = 0; ky < 3; ++ky)
#pragma unroll
            for (int kx = 0; kx < 3; ++kx) {
                int yy = y + ky - 1, xx = x + kx - 1;
                if (yy >= 0 && yy < HQ_ && xx >= 0 && xx < HQ_)
                    acc += wr[ci * 9 + ky * 3 + kx] * grid[(size_t)ci * Q_ + yy * HQ_ + xx];
            }
    wemb[(size_t)p * DIM_ + c] = acc;
}

// img_emb = normalize_c(conv3x3(d) - c_embed) -> (bn,1680,128)
__global__ __launch_bounds__(128) void iemb_kernel(const float* __restrict__ dg,
                                                   const float* __restrict__ cemb,
                                                   const float* __restrict__ iw,
                                                   float* __restrict__ iemb) {
    int idx = blockIdx.x;
    int p = idx % PIX, bn = idx / PIX;
    int y = p / FW_, x = p % FW_, c = threadIdx.x;
    const float* wr = iw + (size_t)c * 36;
    float acc = 0.f;
#pragma unroll
    for (int ci = 0; ci < 4; ++ci)
#pragma unroll
        for (int ky = 0; ky < 3; ++ky)
#pragma unroll
            for (int kx = 0; kx < 3; ++kx) {
                int yy = y + ky - 1, xx = x + kx - 1;
                if (yy >= 0 && yy < FH_ && xx >= 0 && xx < FW_)
                    acc += wr[ci * 9 + ky * 3 + kx] *
                           dg[((size_t)(bn * 4 + ci)) * PIX + yy * FW_ + xx];
            }
    acc -= cemb[bn * DIM_ + c];
    __shared__ float red[128];
    red[c] = acc * acc; __syncthreads();
    for (int st = 64; st > 0; st >>= 1) { if (c < st) red[c] += red[c + st]; __syncthreads(); }
    float nrm = sqrtf(red[0]) + 1e-7f;
    iemb[((size_t)(bn * PIX) + p) * DIM_ + c] = acc / nrm;
}

// q = normalize_c(w_embed - c_embed) + x -> q_bf (bn,2500,128) & q_bf2 (b,2500,768)
__global__ __launch_bounds__(128) void qpos_kernel(const float* __restrict__ wemb,
                                                   const float* __restrict__ cemb,
                                                   const float* __restrict__ x,
                                                   __bf16* __restrict__ qb,
                                                   __bf16* __restrict__ qb2) {
    int idx = blockIdx.x;
    int p = idx % Q_, bn = idx / Q_;
    int b = bn / NCAM, n = bn % NCAM, c = threadIdx.x;
    float e = wemb[(size_t)p * DIM_ + c] - cemb[bn * DIM_ + c];
    __shared__ float red[128];
    red[c] = e * e; __syncthreads();
    for (int st = 64; st > 0; st >>= 1) { if (c < st) red[c] += red[c + st]; __syncthreads(); }
    float nrm = sqrtf(red[0]) + 1e-7f;
    float qv = e / nrm + x[((size_t)(b * DIM_ + c)) * Q_ + p];
    qb[((size_t)(bn * Q_) + p) * DIM_ + c] = (__bf16)qv;
    qb2[((size_t)(b * Q_) + p) * (NCAM * DIM_) + n * DIM_ + c] = (__bf16)qv;
}

// fused BN(eval)+ReLU + NCHW->NHWC transpose -> bf16 xh2 (bn, pix, cin)
__global__ void bnrelu_t_kernel(const float* __restrict__ f, const float* __restrict__ g,
                                const float* __restrict__ bb, const float* __restrict__ rm,
                                const float* __restrict__ rv, __bf16* __restrict__ xh2) {
    int idx = blockIdx.x * 256 + threadIdx.x;
    if (idx >= BN * PIX * FEAT) return;
    int cin = idx % FEAT;
    int p   = (idx / FEAT) % PIX;
    int bn  = idx / (FEAT * PIX);
    float sc = g[cin] * rsqrtf(rv[cin] + 1e-5f);
    float v = (f[((size_t)bn * FEAT + cin) * PIX + p] - rm[cin]) * sc + bb[cin];
    xh2[idx] = (__bf16)fmaxf(v, 0.f);
}

// pack weight W(out_f x in_f, f32, row-major over in) into WMMA-B fragment order:
// frag[kc][ntile][lane][e] = W[ntile*16 + lane%16, kc*32 + (lane/16)*16 + e]
__global__ void pack_wfrag_kernel(const float* __restrict__ w, __bf16* __restrict__ frag,
                                  int out_f, int in_f) {
    int idx = blockIdx.x * 256 + threadIdx.x;
    int total = (in_f >> 5) * (out_f >> 4) * 512;
    if (idx >= total) return;
    int e = idx & 15;
    int lane = (idx >> 4) & 31;
    int rest = idx >> 9;
    int ntile = rest % (out_f >> 4);
    int kc = rest / (out_f >> 4);
    int k = kc * 32 + (lane >> 4) * 16 + e;
    int n = ntile * 16 + (lane & 15);
    frag[idx] = (__bf16)w[(size_t)n * in_f + k];
}

// conv weight (128,256,3,3) into fragments with TAP-MAJOR K: k = tap*256 + cin
__global__ void pack_wfrag_conv_kernel(const float* __restrict__ w, __bf16* __restrict__ frag) {
    int idx = blockIdx.x * 256 + threadIdx.x;
    if (idx >= KCCH * 8 * 512) return;
    int e = idx & 15;
    int lane = (idx >> 4) & 31;
    int rest = idx >> 9;
    int ntile = rest & 7;
    int kc = rest >> 3;
    int k = kc * 32 + (lane >> 4) * 16 + e;
    int tap = k >> 8, cin = k & 255;
    int n = ntile * 16 + (lane & 15);
    frag[idx] = (__bf16)w[((size_t)n * FEAT + cin) * 9 + tap];
}

// pack V (bn,784,128 bf16) into B-fragments with zero-padded tail keys
__global__ void pack_vfrag_kernel(const __bf16* __restrict__ vv, __bf16* __restrict__ frag) {
    int idx = blockIdx.x * 256 + threadIdx.x;
    if (idx >= BN * KCH * 8 * 512) return;
    int e = idx & 15;
    int lane = (idx >> 4) & 31;
    int dt = (idx >> 9) & 7;
    int rest = idx >> 12;
    int kc = rest % KCH;
    int bn = rest / KCH;
    int k = kc * 32 + (lane >> 4) * 16 + e;
    int col = dt * 16 + (lane & 15);
    __bf16 v = (__bf16)0.0f;
    if (k < KKEYS) v = vv[((size_t)bn * KKEYS + k) * DIM_ + col];
    frag[idx] = v;
}

// ===========================================================================
// conv3x3 256->128 implicit GEMM, tap-major K, channel-last activations.
// Per wave: 16 pixels x 32 out-channels; A = two b128 loads per K-step.
__global__ __launch_bounds__(128) void conv_kernel(const __bf16* __restrict__ xh2,
                                                   const __bf16* __restrict__ wf,
                                                   const float* __restrict__ iemb,
                                                   float* __restrict__ outp) {
    int lane = threadIdx.x & 31, warp = threadIdx.x >> 5;
    int tile = blockIdx.x * 4 + warp;
    if (tile >= BN * 4 * 105) return;               // wave-uniform
    int pt = tile % 105; tile /= 105;
    int ct2 = tile % 4;  int img = tile / 4;        // channels [ct2*32, ct2*32+32)
    int half = lane >> 4, l16 = lane & 15;
    int pix = pt * 16 + l16;
    int y = pix / FW_, x = pix % FW_;
    const __bf16* ibase = xh2 + (size_t)img * PIX * FEAT;
    v8f acc0 = {}, acc1 = {};
#pragma unroll
    for (int tap = 0; tap < 9; ++tap) {
        int yy = y + tap / 3 - 1, xx = x + tap % 3 - 1;
        bool inb = (yy >= 0 && yy < FH_ && xx >= 0 && xx < FW_);
        const __bf16* rp = ibase + ((size_t)(yy * FW_ + xx)) * FEAT;
#pragma unroll
        for (int c8 = 0; c8 < 8; ++c8) {
            int kc = tap * 8 + c8;
            v16bf A = {};
            if (inb) {
                const __bf16* ap = rp + c8 * 32;
                A = ld16pair(ap + half * 16, ap + half * 16 + 8);
            }
            const __bf16* bp = wf + (((size_t)kc * 8 + ct2 * 2) * 32 + lane) * 16;
            v16bf b0 = ld16pair(bp, bp + 8);
            v16bf b1 = ld16pair(bp + 512, bp + 520);
            if (c8 == 0)
                __builtin_prefetch(wf + (((size_t)kc * 8 + ct2 * 2) * 32) * 16 + 8192, 0, 1);
            acc0 = WMMA_BF16(A, b0, acc0);
            acc1 = WMMA_BF16(A, b1, acc1);
        }
    }
#pragma unroll
    for (int i = 0; i < 8; ++i) {
        int prow = pt * 16 + half * 8 + i;
        size_t ix = ((size_t)img * PIX + prow) * DIM_ + ct2 * 32 + l16;
        float v0 = acc0[i], v1 = acc1[i];
        if (iemb) { v0 += iemb[ix]; v1 += iemb[ix + 16]; }
        outp[ix] = v0; outp[ix + 16] = v1;
    }
}

// adaptive pool (28x60 -> 28x28) -> bf16 (bn,784,128)
__global__ __launch_bounds__(128) void pool_kernel(const float* __restrict__ key,
                                                   const float* __restrict__ val,
                                                   __bf16* __restrict__ kp,
                                                   __bf16* __restrict__ vp) {
    int idx = blockIdx.x;
    int op = idx % KKEYS, bn = idx / KKEYS;
    int oy = op / MS, ox = op % MS;
    int xs = (ox * FW_) / MS, xe = ((ox + 1) * FW_ + MS - 1) / MS;
    float inv = 1.0f / (float)(xe - xs);
    int c = threadIdx.x;
    float sk = 0.f, sv = 0.f;
    for (int xx = xs; xx < xe; ++xx) {
        size_t i = ((size_t)(bn * PIX) + oy * FW_ + xx) * DIM_ + c;
        sk += key[i]; sv += val[i];
    }
    kp[(size_t)idx * DIM_ + c] = (__bf16)(sk * inv);
    vp[(size_t)idx * DIM_ + c] = (__bf16)(sv * inv);
}

// ===========================================================================
// GEMM: C[MxN] = A[MxK](bf16,row-major) @ Bfrag + bias (+addsrc); 16x32 per wave
__global__ __launch_bounds__(128) void gemm_wmma(const __bf16* __restrict__ A,
                                                 const __bf16* __restrict__ Bf,
                                                 const float* __restrict__ bias,
                                                 const float* __restrict__ addsrc,
                                                 float* __restrict__ Cf,
                                                 __bf16* __restrict__ Cbf,
                                                 int M, int N, int K) {
    int lane = threadIdx.x & 31, warp = threadIdx.x >> 5;
    int mt = (M + 15) >> 4, nt2 = N >> 5;
    long tile = (long)blockIdx.x * 4 + warp;
    if (tile >= (long)mt * nt2) return;             // wave-uniform
    int tn = (int)(tile % nt2), tm = (int)(tile / nt2);
    int half = lane >> 4, l16 = lane & 15;
    int row = tm * 16 + l16; if (row >= M) row = M - 1;
    int kch = K >> 5, ntf = N >> 4;
    v8f acc0 = {}, acc1 = {};
    for (int kc = 0; kc < kch; ++kc) {
        const __bf16* ap = A + (size_t)row * K + kc * 32;
        v16bf a = ld16pair(ap + half * 16, ap + half * 16 + 8);
        const __bf16* bp = Bf + (((size_t)kc * ntf + tn * 2) * 32 + lane) * 16;
        v16bf b0 = ld16pair(bp, bp + 8);
        v16bf b1 = ld16pair(bp + 512, bp + 520);
        acc0 = WMMA_BF16(a, b0, acc0);
        acc1 = WMMA_BF16(a, b1, acc1);
    }
    int col = tn * 32 + l16;
    float bv0 = bias ? bias[col] : 0.f;
    float bv1 = bias ? bias[col + 16] : 0.f;
#pragma unroll
    for (int i = 0; i < 8; ++i) {
        int r = tm * 16 + half * 8 + i;
        if (r < M) {
            size_t ix = (size_t)r * N + col;
            float v0 = acc0[i] + bv0, v1 = acc1[i] + bv1;
            if (addsrc) { v0 += addsrc[ix]; v1 += addsrc[ix + 16]; }
            if (Cf)  { Cf[ix] = v0;  Cf[ix + 16] = v1; }
            if (Cbf) { Cbf[ix] = (__bf16)v0; Cbf[ix + 16] = (__bf16)v1; }
        }
    }
}

// ===========================================================================
// flash multi-camera joint-softmax attention; one wave per (b, head, 16-query tile)
__global__ __launch_bounds__(32) void attn_kernel(const __bf16* __restrict__ qq,
                                                  const __bf16* __restrict__ kk,
                                                  const __bf16* __restrict__ vfrag,
                                                  float* __restrict__ xo_pre) {
    int t = blockIdx.x;
    int qt = t % QTILES; t /= QTILES;
    int h = t % HEADS_;  int b = t / HEADS_;
    int lane = threadIdx.x, half = lane >> 4, l16 = lane & 15;

    __shared__ alignas(16) float  s_s[16][32];
    __shared__ alignas(16) __bf16 s_p[16][32];
    __shared__ float s_scale[16];
    __shared__ float s_l[16];

    const float scl = 0.17677669529663687f;    // 1/sqrt(DH)
    v8f zero = {};
    v8f O[NCAM][2];
#pragma unroll
    for (int c = 0; c < NCAM; ++c) { O[c][0] = zero; O[c][1] = zero; }
    float m_r = -3.0e38f, l_r = 0.0f;          // valid on lanes < 16

    int q0 = qt * 16;
    for (int cam = 0; cam < NCAM; ++cam) {
        const __bf16* qbase = qq + ((size_t)(b * NCAM + cam) * Q_) * DIM_;
        const __bf16* kbase = kk + ((size_t)(b * NCAM + cam) * KKEYS) * DIM_;
        const __bf16* vbase = vfrag + ((size_t)(b * NCAM + cam) * KCH) * 8 * 512;
        v16bf Aq;
        {
            int r = q0 + l16; if (r > Q_ - 1) r = Q_ - 1;
            const __bf16* rp = qbase + (size_t)r * DIM_ + h * DH_;
            Aq = ld16pair(rp + half * 16, rp + half * 16 + 8);
        }
        for (int kc = 0; kc < KCH; ++kc) {
            int k0 = kc * 32;
            int kvalid = KKEYS - k0; if (kvalid > 32) kvalid = 32;
            v16bf Bk0, Bk1;
            {
                int r0 = k0 + l16;      if (r0 > KKEYS - 1) r0 = KKEYS - 1;
                int r1 = k0 + 16 + l16; if (r1 > KKEYS - 1) r1 = KKEYS - 1;
                const __bf16* rp0 = kbase + (size_t)r0 * DIM_ + h * DH_ + half * 16;
                const __bf16* rp1 = kbase + (size_t)r1 * DIM_ + h * DH_ + half * 16;
                Bk0 = ld16pair(rp0, rp0 + 8);
                Bk1 = ld16pair(rp1, rp1 + 8);
            }
            v8f S0 = zero, S1 = zero;
            S0 = WMMA_BF16(Aq, Bk0, S0);
            S1 = WMMA_BF16(Aq, Bk1, S1);
            __syncthreads();
#pragma unroll
            for (int i = 0; i < 8; ++i) {
                s_s[half * 8 + i][l16]      = S0[i] * scl;
                s_s[half * 8 + i][16 + l16] = S1[i] * scl;
            }
            __syncthreads();
            if (lane < 16) {
                float mc = -3.0e38f;
                for (int j = 0; j < 32; ++j) {
                    float sv = (j < kvalid) ? s_s[lane][j] : -3.0e38f;
                    mc = fmaxf(mc, sv);
                }
                float mnew = fmaxf(m_r, mc);
                float sc = __expf(m_r - mnew);
                float ssum = 0.f;
                for (int j = 0; j < 32; ++j) {
                    float p = (j < kvalid) ? __expf(s_s[lane][j] - mnew) : 0.f;
                    s_p[lane][j] = (__bf16)p;
                    ssum += p;
                }
                l_r = l_r * sc + ssum; m_r = mnew;
                s_scale[lane] = sc;
            }
            __syncthreads();
#pragma unroll
            for (int i = 0; i < 8; ++i) {
                float f = s_scale[half * 8 + i];
#pragma unroll
                for (int c2 = 0; c2 < NCAM; ++c2) { O[c2][0][i] *= f; O[c2][1][i] *= f; }
            }
            const __bf16* pp = &s_p[l16][0];
            v16bf Pa = ld16pair(pp + half * 16, pp + half * 16 + 8);
            const __bf16* vfr = vbase + (((size_t)kc * 8) + h * 2) * 512 + lane * 16;
            v16bf Bv0 = ld16pair(vfr, vfr + 8);
            v16bf Bv1 = ld16pair(vfr + 512, vfr + 520);
            O[cam][0] = WMMA_BF16(Pa, Bv0, O[cam][0]);
            O[cam][1] = WMMA_BF16(Pa, Bv1, O[cam][1]);
        }
    }
    if (lane < 16) s_l[lane] = l_r;
    __syncthreads();
#pragma unroll
    for (int i = 0; i < 8; ++i) {
        int row = q0 + half * 8 + i;
        if (row < Q_) {
            float inv = 1.0f / s_l[half * 8 + i];
            size_t base = ((size_t)(b * Q_ + row)) * (NCAM * DIM_);
#pragma unroll
            for (int c2 = 0; c2 < NCAM; ++c2) {
                xo_pre[base + c2 * DIM_ + h * DH_ + l16]      = O[c2][0][i] * inv;
                xo_pre[base + c2 * DIM_ + h * DH_ + 16 + l16] = O[c2][1][i] * inv;
            }
        }
    }
}

// ===========================================================================
__global__ __launch_bounds__(256) void ln768_kernel(const float* __restrict__ xin,
                                                    const float* __restrict__ g,
                                                    const float* __restrict__ bta,
                                                    __bf16* __restrict__ out) {
    int row = blockIdx.x;
    const float* xr = xin + (size_t)row * 768;
    __shared__ float red[256];
    float s = 0.f;
    for (int j = threadIdx.x; j < 768; j += 256) s += xr[j];
    red[threadIdx.x] = s; __syncthreads();
    for (int st = 128; st > 0; st >>= 1) { if (threadIdx.x < st) red[threadIdx.x] += red[threadIdx.x + st]; __syncthreads(); }
    float mu = red[0] / 768.f; __syncthreads();
    float s2 = 0.f;
    for (int j = threadIdx.x; j < 768; j += 256) { float d = xr[j] - mu; s2 += d * d; }
    red[threadIdx.x] = s2; __syncthreads();
    for (int st = 128; st > 0; st >>= 1) { if (threadIdx.x < st) red[threadIdx.x] += red[threadIdx.x + st]; __syncthreads(); }
    float rstd = rsqrtf(red[0] / 768.f + 1e-5f);
    for (int j = threadIdx.x; j < 768; j += 256)
        out[(size_t)row * 768 + j] = (__bf16)((xr[j] - mu) * rstd * g[j] + bta[j]);
}

__global__ void gelu_bf_kernel(const float* __restrict__ src, __bf16* __restrict__ dst, int n) {
    int i = blockIdx.x * 256 + threadIdx.x;
    if (i >= n) return;
    float v = src[i];
    dst[i] = (__bf16)(0.5f * v * (1.0f + erff(v * 0.70710678118654752f)));
}

// out[b,c,p] = xo[b,p,c] + LN_128(h2[b,p,:])[c]
__global__ __launch_bounds__(128) void final_kernel(const float* __restrict__ xo,
                                                    const float* __restrict__ h2,
                                                    const float* __restrict__ g,
                                                    const float* __restrict__ bta,
                                                    float* __restrict__ out) {
    int row = blockIdx.x;
    int b = row / Q_, p = row % Q_;
    int c = threadIdx.x;
    __shared__ float red[128];
    float v = h2[(size_t)row * DIM_ + c];
    red[c] = v; __syncthreads();
    for (int st = 64; st > 0; st >>= 1) { if (c < st) red[c] += red[c + st]; __syncthreads(); }
    float mu = red[0] / 128.f; __syncthreads();
    float d = v - mu;
    red[c] = d * d; __syncthreads();
    for (int st = 64; st > 0; st >>= 1) { if (c < st) red[c] += red[c + st]; __syncthreads(); }
    float rstd = rsqrtf(red[0] / 128.f + 1e-5f);
    out[((size_t)(b * DIM_ + c)) * Q_ + p] = xo[(size_t)row * DIM_ + c] + d * rstd * g[c] + bta[c];
}

// ===========================================================================
extern "C" void kernel_launch(void* const* d_in, const int* in_sizes, int n_in,
                              void* d_out, int out_size, void* d_ws, size_t ws_size,
                              hipStream_t stream) {
    const float* x       = (const float*)d_in[0];
    const float* feature = (const float*)d_in[1];
    const float* I_inv   = (const float*)d_in[2];
    const float* E_inv   = (const float*)d_in[3];
    const float* plane   = (const float*)d_in[4];
    const float* bevgrid = (const float*)d_in[5];
    const float* fl_g  = (const float*)d_in[6];
    const float* fl_b  = (const float*)d_in[7];
    const float* fl_rm = (const float*)d_in[8];
    const float* fl_rv = (const float*)d_in[9];
    const float* fl_w  = (const float*)d_in[10];
    const float* fp_g  = (const float*)d_in[11];
    const float* fp_b  = (const float*)d_in[12];
    const float* fp_rm = (const float*)d_in[13];
    const float* fp_rv = (const float*)d_in[14];
    const float* fp_w  = (const float*)d_in[15];
    const float* bev_w = (const float*)d_in[16];
    const float* img_w = (const float*)d_in[17];
    const float* cam_w = (const float*)d_in[18];
    const float* wq_w  = (const float*)d_in[19];
    const float* wq_b  = (const float*)d_in[20];
    const float* wk_w  = (const float*)d_in[21];
    const float* wk_b  = (const float*)d_in[22];
    const float* wv_w  = (const float*)d_in[23];
    const float* wv_b  = (const float*)d_in[24];
    const float* addq_w = (const float*)d_in[25];
    const float* addq_b = (const float*)d_in[26];
    const float* pre_g  = (const float*)d_in[27];
    const float* pre_b  = (const float*)d_in[28];
    const float* proj_w = (const float*)d_in[29];
    const float* proj_b = (const float*)d_in[30];
    const float* nrm_g  = (const float*)d_in[31];
    const float* nrm_b  = (const float*)d_in[32];
    const float* m1_w   = (const float*)d_in[33];
    const float* m1_b   = (const float*)d_in[34];
    const float* m2_w   = (const float*)d_in[35];
    const float* m2_b   = (const float*)d_in[36];
    float* out = (float*)d_out;

    char* wsp = (char*)d_ws;
    size_t off = 0;
    auto alloc = [&](size_t bytes) -> void* {
        void* p = wsp + off;
        off += (bytes + 255) & ~(size_t)255;
        return p;
    };
    float*  dg    = (float*)alloc((size_t)BN * 4 * PIX * 4);
    float*  cemb  = (float*)alloc((size_t)BN * DIM_ * 4);
    float*  wemb  = (float*)alloc((size_t)Q_ * DIM_ * 4);
    float*  iemb  = (float*)alloc((size_t)BN * PIX * DIM_ * 4);
    __bf16* xh2   = (__bf16*)alloc((size_t)BN * PIX * FEAT * 2);
    __bf16* wflf  = (__bf16*)alloc((size_t)KCONV * DIM_ * 2);
    __bf16* wfpf  = (__bf16*)alloc((size_t)KCONV * DIM_ * 2);
    float*  keyb  = (float*)alloc((size_t)BN * PIX * DIM_ * 4);
    float*  valb  = (float*)alloc((size_t)BN * PIX * DIM_ * 4);
    __bf16* kpb   = (__bf16*)alloc((size_t)BN * KKEYS * DIM_ * 2);
    __bf16* vpb   = (__bf16*)alloc((size_t)BN * KKEYS * DIM_ * 2);
    __bf16* qb    = (__bf16*)alloc((size_t)BN * Q_ * DIM_ * 2);
    __bf16* qb2   = (__bf16*)alloc((size_t)B_ * Q_ * NCAM * DIM_ * 2);
    __bf16* wqf   = (__bf16*)alloc((size_t)DIM_ * DIM_ * 2);
    __bf16* wkf   = (__bf16*)alloc((size_t)DIM_ * DIM_ * 2);
    __bf16* wvf   = (__bf16*)alloc((size_t)DIM_ * DIM_ * 2);
    __bf16* addqf = (__bf16*)alloc((size_t)NCAM * DIM_ * DIM_ * 2);
    __bf16* projf = (__bf16*)alloc((size_t)NCAM * DIM_ * DIM_ * 2);
    __bf16* m1f   = (__bf16*)alloc((size_t)DIM_ * 2 * DIM_ * 2);
    __bf16* m2f   = (__bf16*)alloc((size_t)2 * DIM_ * DIM_ * 2);
    __bf16* qqb   = (__bf16*)alloc((size_t)BN * Q_ * DIM_ * 2);
    __bf16* kkb   = (__bf16*)alloc((size_t)BN * KKEYS * DIM_ * 2);
    __bf16* vvb   = (__bf16*)alloc((size_t)BN * KKEYS * DIM_ * 2);
    __bf16* vfrag = (__bf16*)alloc((size_t)BN * KCH * 8 * 512 * 2);
    float*  xop   = (float*)alloc((size_t)B_ * Q_ * NCAM * DIM_ * 4);
    __bf16* xoln  = (__bf16*)alloc((size_t)B_ * Q_ * NCAM * DIM_ * 2);
    float*  addqo = (float*)alloc((size_t)B_ * Q_ * DIM_ * 4);
    float*  xof   = (float*)alloc((size_t)B_ * Q_ * DIM_ * 4);
    __bf16* xob   = (__bf16*)alloc((size_t)B_ * Q_ * DIM_ * 2);
    float*  h1f   = (float*)alloc((size_t)B_ * Q_ * 2 * DIM_ * 4);
    __bf16* h1b   = (__bf16*)alloc((size_t)B_ * Q_ * 2 * DIM_ * 2);
    float*  h2f   = (float*)alloc((size_t)B_ * Q_ * DIM_ * 4);
    (void)ws_size; (void)in_sizes; (void)n_in; (void)out_size;

    auto gemm = [&](const __bf16* A, const __bf16* Bf, const float* bias,
                    const float* addsrc, float* Cf, __bf16* Cbf, int M, int N, int K) {
        long tiles = (long)((M + 15) / 16) * (N / 32);
        int blocks = (int)((tiles + 3) / 4);
        gemm_wmma<<<blocks, 128, 0, stream>>>(A, Bf, bias, addsrc, Cf, Cbf, M, N, K);
    };
    auto pack = [&](const float* w, __bf16* frag, int out_f, int in_f) {
        int nel = (in_f / 32) * (out_f / 16) * 512;
        pack_wfrag_kernel<<<(nel + 255) / 256, 256, 0, stream>>>(w, frag, out_f, in_f);
    };

    // --- geometry / embeddings ---
    geom_kernel<<<(BN * PIX + 255) / 256, 256, 0, stream>>>(I_inv, E_inv, plane, dg);
    cemb_kernel<<<BN, 128, 0, stream>>>(cam_w, E_inv, cemb);
    wemb_kernel<<<Q_, 128, 0, stream>>>(bevgrid, bev_w, wemb);
    iemb_kernel<<<BN * PIX, 128, 0, stream>>>(dg, cemb, img_w, iemb);
    qpos_kernel<<<BN * Q_, 128, 0, stream>>>(wemb, cemb, x, qb, qb2);

    // --- weight fragments ---
    pack_wfrag_conv_kernel<<<(KCCH * 8 * 512 + 255) / 256, 256, 0, stream>>>(fp_w, wfpf);
    pack_wfrag_conv_kernel<<<(KCCH * 8 * 512 + 255) / 256, 256, 0, stream>>>(fl_w, wflf);
    pack(wq_w, wqf, DIM_, DIM_);
    pack(wk_w, wkf, DIM_, DIM_);
    pack(wv_w, wvf, DIM_, DIM_);
    pack(addq_w, addqf, DIM_, NCAM * DIM_);
    pack(proj_w, projf, DIM_, NCAM * DIM_);
    pack(m1_w, m1f, 2 * DIM_, DIM_);
    pack(m2_w, m2f, DIM_, 2 * DIM_);

    // --- key path: fp BN -> conv (+img_emb) ---
    bnrelu_t_kernel<<<(BN * PIX * FEAT + 255) / 256, 256, 0, stream>>>(
        feature, fp_g, fp_b, fp_rm, fp_rv, xh2);
    conv_kernel<<<(BN * 4 * 105 + 3) / 4, 128, 0, stream>>>(xh2, wfpf, iemb, keyb);
    // --- value path: fl BN -> conv ---
    bnrelu_t_kernel<<<(BN * PIX * FEAT + 255) / 256, 256, 0, stream>>>(
        feature, fl_g, fl_b, fl_rm, fl_rv, xh2);
    conv_kernel<<<(BN * 4 * 105 + 3) / 4, 128, 0, stream>>>(xh2, wflf, nullptr, valb);

    // --- pooling -> bf16 rows ---
    pool_kernel<<<BN * KKEYS, 128, 0, stream>>>(keyb, valb, kpb, vpb);

    // --- projections (WMMA) ---
    gemm(qb,  wqf, wq_b, nullptr, nullptr, qqb, BN * Q_, DIM_, DIM_);
    gemm(kpb, wkf, wk_b, nullptr, nullptr, kkb, BN * KKEYS, DIM_, DIM_);
    gemm(vpb, wvf, wv_b, nullptr, nullptr, vvb, BN * KKEYS, DIM_, DIM_);
    gemm(qb2, addqf, addq_b, nullptr, addqo, nullptr, B_ * Q_, DIM_, NCAM * DIM_);
    pack_vfrag_kernel<<<(BN * KCH * 8 * 512 + 255) / 256, 256, 0, stream>>>(vvb, vfrag);

    // --- flash attention (joint softmax over cams) ---
    attn_kernel<<<B_ * HEADS_ * QTILES, 32, 0, stream>>>(qqb, kkb, vfrag, xop);

    // --- epilogue ---
    ln768_kernel<<<B_ * Q_, 256, 0, stream>>>(xop, pre_g, pre_b, xoln);
    gemm(xoln, projf, proj_b, addqo, xof, xob, B_ * Q_, DIM_, NCAM * DIM_);
    gemm(xob, m1f, m1_b, nullptr, h1f, nullptr, B_ * Q_, 2 * DIM_, DIM_);
    gelu_bf_kernel<<<(B_ * Q_ * 2 * DIM_ + 255) / 256, 256, 0, stream>>>(
        h1f, h1b, B_ * Q_ * 2 * DIM_);
    gemm(h1b, m2f, m2_b, nullptr, h2f, nullptr, B_ * Q_, DIM_, 2 * DIM_);
    final_kernel<<<B_ * Q_, 128, 0, stream>>>(xof, h2f, nrm_g, nrm_b, out);
}
